// AttentionGAT_4578435138191
// MI455X (gfx1250) — compile-verified, matrix operand order checked by reading, not run on hardware
//
#include <hip/hip_runtime.h>

typedef __attribute__((ext_vector_type(16))) _Float16 v16h;
typedef __attribute__((ext_vector_type(8)))  _Float16 v8h;
typedef __attribute__((ext_vector_type(8)))  float    v8f;

#define NNODES 21
#define HID    32
#define PP     64
#define NEDGE  60
#define ETOT   81    // 60 edges + 21 self loops
#define BATCH  1024
#define FEAT   3
#define ROWS_A 168   // 8 periods x 21 nodes
#define ROWS_PAD 176 // 11 M-tiles of 16
#define SINK_ROW 32  // pooled row used as atomic sink for pad rows

// ---------------------------------------------------------------------------
// CDNA5 16-bit WMMA fragment layout (ISA 7.12.2):
//   A (16x32): lane<16 -> M=lane, K halves = {kb..kb+7, kb+16..kb+23}, kb=0
//              lane>=16 -> M=lane-16, kb=8
//   B (32x16): same K mapping with N=lane&15 (+16*ntile)
//   C/D v8f:   elem i -> M = i + (lane<16?0:8), N = lane&15 (+16*ntile)
// A fragment == two contiguous 8-half (16B) runs -> two ds_load_b128.
// ---------------------------------------------------------------------------
__device__ __forceinline__ v16h loadA_lds(const _Float16* G, int r, bool hi) {
  const _Float16* row = G + r * HID;
  v8h lo = *(const v8h*)(row + (hi ? 8 : 0));
  v8h hh = *(const v8h*)(row + (hi ? 24 : 16));
  return __builtin_shufflevector(lo, hh, 0, 1, 2, 3, 4, 5, 6, 7,
                                 8, 9, 10, 11, 12, 13, 14, 15);
}

// B fragments pre-packed lane-major by the prep kernel: 32B per lane.
__device__ __forceinline__ v16h loadB_pack(const _Float16* Bp, int nt, int lane) {
  const _Float16* p = Bp + ((size_t)(nt * 32 + lane)) * 16;
  v8h lo = *(const v8h*)(p);
  v8h hh = *(const v8h*)(p + 8);
  return __builtin_shufflevector(lo, hh, 0, 1, 2, 3, 4, 5, 6, 7,
                                 8, 9, 10, 11, 12, 13, 14, 15);
}

// Branch-free activations (native exp + rcp; saturate correctly via rcp(inf)=0)
__device__ __forceinline__ float fast_sigmoid(float x) {
  return __builtin_amdgcn_rcpf(1.f + __expf(-x));
}
__device__ __forceinline__ float fast_tanh(float x) {
  return 1.f - 2.f * __builtin_amdgcn_rcpf(__expf(2.f * x) + 1.f);
}

// ---------------------------------------------------------------------------
// Prep kernel: GCN norms, softmax(att), and f16 B-fragment packing.
// Pack layout: Bp[(nt*32 + lane)*16 + h], h<8 -> K=kb_lo+h, h>=8 -> K=kb_hi+h-8
// ---------------------------------------------------------------------------
__global__ void a3t_prep_kernel(const int* __restrict__ ei, const float* __restrict__ att,
                                const float* __restrict__ lzW, const float* __restrict__ lhW,
                                const float* __restrict__ l1W,
                                _Float16* __restrict__ LzP, _Float16* __restrict__ LhP,
                                _Float16* __restrict__ L1P,
                                float* __restrict__ norm, int* __restrict__ sArr,
                                int* __restrict__ dArr, float* __restrict__ probs) {
  __shared__ float deg[NNODES];
  __shared__ float dinv[NNODES];
  const int tid = threadIdx.x;
  if (tid < NNODES) deg[tid] = 1.0f;  // self loop
  __syncthreads();
  if (tid == 0) {
    for (int i = 0; i < NEDGE; ++i) deg[ei[NEDGE + i]] += 1.0f;
  }
  __syncthreads();
  if (tid < NNODES) dinv[tid] = (deg[tid] > 0.f) ? rsqrtf(deg[tid]) : 0.f;
  __syncthreads();
  for (int e = tid; e < ETOT; e += blockDim.x) {
    int s = (e < NEDGE) ? ei[e] : (e - NEDGE);
    int d = (e < NEDGE) ? ei[NEDGE + e] : (e - NEDGE);
    sArr[e] = s; dArr[e] = d;
    norm[e] = dinv[s] * dinv[d];
  }
  if (tid == 0) {
    float m = att[0];
    for (int i = 1; i < PP; ++i) m = fmaxf(m, att[i]);
    float tmp[PP];
    float sum = 0.f;
    for (int i = 0; i < PP; ++i) { tmp[i] = __expf(att[i] - m); sum += tmp[i]; }
    float inv = 1.f / sum;
    for (int i = 0; i < PP; ++i) probs[i] = tmp[i] * inv;
  }
  // Pack B fragments. Element index: (nt, lane, h) flat.
  // lz_W / lh_W: rows 0..31 only (H==0 kills bottom half); ncols = 32.
  for (int idx = tid; idx < 2 * 32 * 16; idx += blockDim.x) {
    int h    = idx & 15;
    int lane = (idx >> 4) & 31;
    int nt   = idx >> 9;
    int kb   = (lane >= 16) ? 8 : 0;
    int k    = (h < 8) ? (kb + h) : (kb + 16 + (h - 8));
    int ncol = (lane & 15) + nt * 16;
    LzP[idx] = (_Float16)lzW[k * HID + ncol];
    LhP[idx] = (_Float16)lhW[k * HID + ncol];
  }
  // l1_W: 32x64, 4 n-tiles.
  for (int idx = tid; idx < 4 * 32 * 16; idx += blockDim.x) {
    int h    = idx & 15;
    int lane = (idx >> 4) & 31;
    int nt   = idx >> 9;
    int kb   = (lane >= 16) ? 8 : 0;
    int k    = (h < 8) ? (kb + h) : (kb + 16 + (h - 8));
    int ncol = (lane & 15) + nt * 16;
    L1P[idx] = (_Float16)l1W[k * PP + ncol];
  }
}

// ---------------------------------------------------------------------------
// Fused A3TGCN forward: one block per batch element b (256 threads, 8 waves).
// ---------------------------------------------------------------------------
__global__ __launch_bounds__(256)
void a3tgcn_fused_kernel(const float* __restrict__ x,
                         const float* __restrict__ Wz, const float* __restrict__ bz,
                         const float* __restrict__ Wh, const float* __restrict__ bh,
                         const float* __restrict__ lzb, const float* __restrict__ lhb,
                         const float* __restrict__ l1b,
                         const float* __restrict__ l2W, const float* __restrict__ l2b,
                         const float* __restrict__ norm, const int* __restrict__ sArr,
                         const int* __restrict__ dArr, const float* __restrict__ probs,
                         const _Float16* __restrict__ LzP, const _Float16* __restrict__ LhP,
                         const _Float16* __restrict__ L1P,
                         float* __restrict__ out) {
  __shared__ _Float16 Gz[ROWS_PAD * HID];   // padded to 11 tiles; pad rows zero
  __shared__ _Float16 Gh[ROWS_PAD * HID];
  __shared__ float Xs[8 * NNODES * FEAT];
  __shared__ float Agg[8 * NNODES * FEAT];
  __shared__ float pooled[33 * HID];        // rows 0..20 real, 21..31 zero pad
                                            // (Phase B A-tiles), row 32 = sink
  __shared__ float tbuf[NNODES * PP];
  __shared__ float nE[ETOT];
  __shared__ int   sE[ETOT];
  __shared__ int   dE[ETOT];
  __shared__ float prS[PP];

  const int b    = blockIdx.x;
  const int tid  = threadIdx.x;
  const int wave = tid >> 5;
  const int lane = tid & 31;
  const bool hi  = lane >= 16;
  const int M0   = lane & 15;

  // ---- one-time block init ----
  for (int i = tid; i < 33 * HID; i += 256) pooled[i] = 0.f;
  for (int i = tid; i < ETOT; i += 256) { nE[i] = norm[i]; sE[i] = sArr[i]; dE[i] = dArr[i]; }
  if (tid < PP) prS[tid] = probs[tid];
  // zero the G pad rows (168..175) once; never written afterwards
  for (int i = tid; i < (ROWS_PAD - ROWS_A) * HID; i += 256) {
    Gz[ROWS_A * HID + i] = (_Float16)0.f;
    Gh[ROWS_A * HID + i] = (_Float16)0.f;
  }

  // ---- loop-invariant WMMA operands ----
  const v16h bZ0 = loadB_pack(LzP, 0, lane);
  const v16h bZ1 = loadB_pack(LzP, 1, lane);
  const v16h bH0 = loadB_pack(LhP, 0, lane);
  const v16h bH1 = loadB_pack(LhP, 1, lane);
  const float lzb0 = lzb[M0], lzb1 = lzb[M0 + 16];
  const float lhb0 = lhb[M0], lhb1 = lhb[M0 + 16];
  __syncthreads();

  // ---- Phase A: 8 chunks of 8 periods; wave w owns period c*8+w ----
  for (int c = 0; c < 8; ++c) {
    const int p = c * 8 + wave;

    for (int idx = lane; idx < NNODES * FEAT; idx += 32) {
      int n = idx / FEAT, f = idx - n * FEAT;
      Xs[wave * 63 + idx]  = x[(((size_t)b * NNODES + n) * FEAT + f) * PP + p];
      Agg[wave * 63 + idx] = 0.f;
    }
    __syncthreads();

    for (int idx = lane; idx < ETOT * FEAT; idx += 32) {
      int e = idx / FEAT, f = idx - e * FEAT;
      atomicAdd(&Agg[wave * 63 + dE[e] * FEAT + f],
                nE[e] * Xs[wave * 63 + sE[e] * FEAT + f]);
    }
    __syncthreads();

    // G = Agg @ W + b  (lane = output column k), stored f16 for WMMA A
    {
      const int k = lane;
      const float bzk = bz[k], bhk = bh[k];
      const float wz0 = Wz[0 * HID + k], wz1 = Wz[1 * HID + k], wz2 = Wz[2 * HID + k];
      const float wh0 = Wh[0 * HID + k], wh1 = Wh[1 * HID + k], wh2 = Wh[2 * HID + k];
#pragma unroll 3
      for (int n = 0; n < NNODES; ++n) {
        float a0 = Agg[wave * 63 + n * 3 + 0];
        float a1 = Agg[wave * 63 + n * 3 + 1];
        float a2 = Agg[wave * 63 + n * 3 + 2];
        Gz[(wave * NNODES + n) * HID + k] = (_Float16)(bzk + a0 * wz0 + a1 * wz1 + a2 * wz2);
        Gh[(wave * NNODES + n) * HID + k] = (_Float16)(bhk + a0 * wh0 + a1 * wh1 + a2 * wh2);
      }
    }
    __syncthreads();

    // WMMA over 11 padded M-tiles; Z & Ht GEMMs, K=32 in one instruction.
    for (int t = wave; t < 11; t += 8) {            // wave-uniform: EXEC all-1s
      const int r = t * 16 + M0;
      const v16h aZ = loadA_lds(Gz, r, hi);
      const v16h aH = loadA_lds(Gh, r, hi);

      // Per-tile row bookkeeping (independent of nt): one division for the
      // base row, then incremental wrap; pad rows route to the sink row.
      const int base = t * 16 + (hi ? 8 : 0);
      int pl  = base / NNODES;
      int rem = base - pl * NNODES;
      int   rowoff[8];
      float prv[8];
#pragma unroll
      for (int i = 0; i < 8; ++i) {
        bool pad  = (base + i) >= ROWS_A;
        rowoff[i] = (pad ? SINK_ROW : rem) * HID;
        prv[i]    = prS[c * 8 + (pad ? 0 : pl)];
        if (++rem == NNODES) { rem = 0; ++pl; }
      }

#pragma unroll
      for (int nt = 0; nt < 2; ++nt) {
        const int ncol = M0 + nt * 16;
        v8f cZ, cH;
        const float bzv = nt ? lzb1 : lzb0;
        const float bhv = nt ? lhb1 : lhb0;
#pragma unroll
        for (int i = 0; i < 8; ++i) { cZ[i] = bzv; cH[i] = bhv; }
        cZ = __builtin_amdgcn_wmma_f32_16x16x32_f16(false, aZ, false, nt ? bZ1 : bZ0,
                                                    (short)0, cZ, false, false);
        cH = __builtin_amdgcn_wmma_f32_16x16x32_f16(false, aH, false, nt ? bH1 : bH0,
                                                    (short)0, cH, false, false);
#pragma unroll
        for (int i = 0; i < 8; ++i) {
          float z  = fast_sigmoid(cZ[i]);
          float ht = fast_tanh(cH[i]);
          atomicAdd(&pooled[rowoff[i] + ncol], prv[i] * ((1.f - z) * ht));  // Hn
        }
      }
    }
    __syncthreads();
  }

  // ---- Phase B: t = relu(pooled) @ l1_W + l1_b (21x32 @ 32x64), 8 tiles ----
  {
    const int mtile = wave >> 2, nt = wave & 3;     // 2 x 4 tiles, one per wave
    const int r    = mtile * 16 + M0;               // < 32, pad rows are zero
    const int ncol = M0 + nt * 16;
    const int kb   = hi ? 8 : 0;
    v16h aP;
#pragma unroll
    for (int h = 0; h < 8; ++h) {
      aP[h]     = (_Float16)fmaxf(pooled[r * HID + kb + h], 0.f);
      aP[h + 8] = (_Float16)fmaxf(pooled[r * HID + kb + 16 + h], 0.f);
    }
    const v16h bP = loadB_pack(L1P, nt, lane);
    v8f cP;
    const float bv = l1b[ncol];
#pragma unroll
    for (int i = 0; i < 8; ++i) cP[i] = bv;
    cP = __builtin_amdgcn_wmma_f32_16x16x32_f16(false, aP, false, bP, (short)0, cP, false, false);
#pragma unroll
    for (int i = 0; i < 8; ++i) {
      int rr = mtile * 16 + i + (hi ? 8 : 0);
      if (rr < NNODES) tbuf[rr * PP + ncol] = cP[i];
    }
  }
  __syncthreads();

  // ---- Phase C: out[b,j,p] = sum_n l2W[n,j] * t[n,p] + l2b[j] ----
  for (int idx = tid; idx < 6 * PP; idx += 256) {
    int j = idx >> 6;
    int p = idx & 63;
    float acc = l2b[j];
#pragma unroll
    for (int n = 0; n < NNODES; ++n) acc += l2W[n * 6 + j] * tbuf[n * PP + p];
    out[((size_t)b * 6 + j) * PP + p] = acc;
  }
}

// ---------------------------------------------------------------------------
extern "C" void kernel_launch(void* const* d_in, const int* in_sizes, int n_in,
                              void* d_out, int out_size, void* d_ws, size_t ws_size,
                              hipStream_t stream) {
  (void)in_sizes; (void)n_in; (void)out_size; (void)ws_size;
  const float* x   = (const float*)d_in[0];
  const int*   ei  = (const int*)d_in[1];
  const float* Wz  = (const float*)d_in[2];
  const float* bz  = (const float*)d_in[3];
  // d_in[4]/d_in[5] (W_r, b_r) dead: H==0 -> H*R == 0.
  const float* Wh  = (const float*)d_in[6];
  const float* bh  = (const float*)d_in[7];
  const float* lzW = (const float*)d_in[8];
  const float* lzb = (const float*)d_in[9];
  // d_in[10]/d_in[11] (lr_W, lr_b) dead.
  const float* lhW = (const float*)d_in[12];
  const float* lhb = (const float*)d_in[13];
  const float* att = (const float*)d_in[14];
  const float* l1W = (const float*)d_in[15];
  const float* l1b = (const float*)d_in[16];
  const float* l2W = (const float*)d_in[17];
  const float* l2b = (const float*)d_in[18];

  // ws layout: 32B-aligned packed f16 fragments first, then f32/i32 tables.
  _Float16* LzP = (_Float16*)d_ws;              // 1024 halves
  _Float16* LhP = LzP + 1024;                   // 1024 halves
  _Float16* L1P = LhP + 1024;                   // 2048 halves
  float* norm   = (float*)(L1P + 2048);         // 81
  int*   sArr   = (int*)(norm + ETOT);          // 81
  int*   dArr   = sArr + ETOT;                  // 81
  float* probs  = (float*)(dArr + ETOT);        // 64

  a3t_prep_kernel<<<1, 128, 0, stream>>>(ei, att, lzW, lhW, l1W,
                                         LzP, LhP, L1P, norm, sArr, dArr, probs);
  a3tgcn_fused_kernel<<<BATCH, 256, 0, stream>>>(x, Wz, bz, Wh, bh, lzb, lhb, l1b,
                                                 l2W, l2b, norm, sArr, dArr, probs,
                                                 LzP, LhP, L1P, (float*)d_out);
}
